// WikiLM_59416577573386
// MI455X (gfx1250) — matrix operand order repaired and checked by reading, not compiled, not run on hardware
//
#include <hip/hip_runtime.h>
#include <math.h>

// ---------------- model dims ----------------
#define BB 2
#define TT 1024
#define VV 50257
#define DD 768
#define FF 3072
#define LL 6
#define HH 12
#define BT (BB*TT)      // 2048
#define D3 (3*DD)       // 2304

// ---------------- WMMA types ----------------
typedef __attribute__((ext_vector_type(16))) __bf16        v16bf;
typedef __attribute__((ext_vector_type(8)))  float         v8f;
typedef __attribute__((ext_vector_type(8)))  unsigned short v8us;

union FragBF { v16bf v; v8us h[2]; };

static __device__ __forceinline__ unsigned short f2bf(float f) {
    unsigned int u = __float_as_uint(f);
    u += 0x7FFFu + ((u >> 16) & 1u);          // RNE
    return (unsigned short)(u >> 16);
}

// CDNA5 async global->LDS copy (16B per lane), tracked by ASYNCcnt.
static __device__ __forceinline__ void async_ld16(unsigned int lds, const void* g) {
    asm volatile("global_load_async_to_lds_b128 %0, %1, off"
                 :: "v"(lds), "v"(g) : "memory");
}
static __device__ __forceinline__ void wait_async0() {
    asm volatile("s_wait_asynccnt 0x0" ::: "memory");
}
static __device__ __forceinline__ unsigned int lds_addr(const void* p) {
    return (unsigned int)(uintptr_t)p;        // LDS aperture: addr[31:0]
}

// max-reduce across each 16-lane row (matches WMMA C-fragment row striping).
// VALU-only butterfly via v_permlane16_b32 when available.
static __device__ __forceinline__ float rowmax16(float x) {
#if __has_builtin(__builtin_amdgcn_permlane16)
    unsigned u;
    u = __builtin_amdgcn_permlane16(__float_as_uint(x), __float_as_uint(x),
                                    0x67452301u, 0xEFCDAB89u, false, false);  // xor 1
    x = fmaxf(x, __uint_as_float(u));
    u = __builtin_amdgcn_permlane16(__float_as_uint(x), __float_as_uint(x),
                                    0x54761032u, 0xDCFE98BAu, false, false);  // xor 2
    x = fmaxf(x, __uint_as_float(u));
    u = __builtin_amdgcn_permlane16(__float_as_uint(x), __float_as_uint(x),
                                    0x32107654u, 0xBA98FEDCu, false, false);  // xor 4
    x = fmaxf(x, __uint_as_float(u));
    u = __builtin_amdgcn_permlane16(__float_as_uint(x), __float_as_uint(x),
                                    0xFEDCBA98u, 0x76543210u, false, false);  // xor 8
    x = fmaxf(x, __uint_as_float(u));
#else
#pragma unroll
    for (int mm = 1; mm < 16; mm <<= 1) x = fmaxf(x, __shfl_xor(x, mm, 16));
#endif
    return x;
}

// ---------------- elementwise kernels ----------------
__global__ void cvt_f32_bf16(const float* __restrict__ in, unsigned short* __restrict__ out, long n) {
    long i = (long)blockIdx.x * blockDim.x + threadIdx.x;
    long stride = (long)gridDim.x * blockDim.x;
    for (; i < n; i += stride) out[i] = f2bf(in[i]);
}

__global__ void embed_kernel(const int* __restrict__ idx, const float* __restrict__ tok,
                             const float* __restrict__ pos, float* __restrict__ x) {
    int bt = blockIdx.x;
    int t = bt % TT;
    int id = idx[bt];
    for (int d = threadIdx.x; d < DD; d += blockDim.x)
        x[(size_t)bt * DD + d] = tok[(size_t)id * DD + d] + pos[(size_t)t * DD + d];
}

// LayerNorm: f32 input row -> bf16 output row (GEMM A operand)
__global__ __launch_bounds__(256) void ln_bf16_kernel(const float* __restrict__ x,
                                                      const float* __restrict__ w,
                                                      const float* __restrict__ b,
                                                      unsigned short* __restrict__ out) {
    int row = blockIdx.x;
    const float* xr = x + (size_t)row * DD;
    float s = 0.f, s2 = 0.f;
    for (int i = threadIdx.x; i < DD; i += 256) { float v = xr[i]; s += v; s2 += v * v; }
#pragma unroll
    for (int m = 16; m; m >>= 1) { s += __shfl_xor(s, m, 32); s2 += __shfl_xor(s2, m, 32); }
    __shared__ float wsum[8], wsq[8];
    int wave = threadIdx.x >> 5, lane = threadIdx.x & 31;
    if (lane == 0) { wsum[wave] = s; wsq[wave] = s2; }
    __syncthreads();
    float ts = 0.f, ts2 = 0.f;
#pragma unroll
    for (int i = 0; i < 8; i++) { ts += wsum[i]; ts2 += wsq[i]; }
    float mean = ts * (1.0f / DD);
    float var = ts2 * (1.0f / DD) - mean * mean;
    float inv = rsqrtf(var + 1e-5f);
    for (int i = threadIdx.x; i < DD; i += 256)
        out[(size_t)row * DD + i] = f2bf((xr[i] - mean) * inv * w[i] + b[i]);
}

// ---------------- bf16 WMMA GEMM ----------------
// C[M,N] = A[M,K] * Bw[N,K]^T (+bias)(+gelu)(+resid); f32 and/or bf16 out.
// Block: 256 thr (8 waves), 128x256 C tile; wave = 64x64 (4x4 WMMA tiles).
// K chunks of 32, double-buffered LDS fed by async global->LDS copies.
// Requires: M % 128 == 0 (true: M = BT = 2048), K % 64 == 0 (768, 3072).
#define GBM 128
#define GBN 256
#define LDA 40   // LDS row stride (elems): 80B, 16B aligned, conflict-free frag reads

template<bool HASBIAS, bool GELU, bool HASRES, bool OUTFP, bool OUTBF, bool FULLN>
__global__ __launch_bounds__(256) __attribute__((amdgpu_waves_per_eu(2)))
void gemm_bf16_wmma(const unsigned short* __restrict__ A,
                    const unsigned short* __restrict__ Bw,
                    const float* __restrict__ bias,
                    const float* __restrict__ resid,
                    float* __restrict__ outF,
                    unsigned short* __restrict__ outB,
                    int M, int N, int K, int bn0)
{
    __shared__ __align__(16) unsigned short As[2][GBM * LDA];
    __shared__ __align__(16) unsigned short Bs[2][GBN * LDA];

    const int bm = blockIdx.y, bn = blockIdx.x + bn0;
    const int tid = threadIdx.x, lane = tid & 31, wave = tid >> 5;
    const int wm = wave >> 2, wn = wave & 3;           // 2x4 wave grid
    const int r = lane & 15, hi = lane >> 4;

    // per-thread chunk mapping (16B chunks: 8 bf16). A: 2 chunks, B: 4 chunks.
    int arow[2], acol[2]; const unsigned short* aG[2]; unsigned aL[2][2];
#pragma unroll
    for (int t = 0; t < 2; t++) {
        int c = tid + t * 256;
        arow[t] = c >> 2; acol[t] = (c & 3) * 8;
        aG[t] = A + (size_t)(bm * GBM + arow[t]) * K + acol[t];
        aL[t][0] = lds_addr(&As[0][arow[t] * LDA + acol[t]]);
        aL[t][1] = lds_addr(&As[1][arow[t] * LDA + acol[t]]);
    }
    int brow[4], bcol[4]; const unsigned short* bG[4]; unsigned bL[4][2];
#pragma unroll
    for (int t = 0; t < 4; t++) {
        int c = tid + t * 256;
        brow[t] = c >> 2; bcol[t] = (c & 3) * 8;
        bG[t] = Bw + (size_t)(bn * GBN + brow[t]) * K + bcol[t];
        bL[t][0] = lds_addr(&Bs[0][brow[t] * LDA + bcol[t]]);
        bL[t][1] = lds_addr(&Bs[1][brow[t] * LDA + bcol[t]]);
    }

    v8f acc[4][4] = {};

    auto load_tiles = [&](int k0, int buf) {
#pragma unroll
        for (int t = 0; t < 2; t++) async_ld16(aL[t][buf], aG[t] + k0);
        if (FULLN) {
#pragma unroll
            for (int t = 0; t < 4; t++) async_ld16(bL[t][buf], bG[t] + k0);
        } else {
#pragma unroll
            for (int t = 0; t < 4; t++) {
                int gn = bn * GBN + brow[t];
                ushort4 bv  = make_ushort4(0, 0, 0, 0);
                ushort4 bv2 = make_ushort4(0, 0, 0, 0);
                if (gn < N) {
                    bv  = *(const ushort4*)(bG[t] + k0);
                    bv2 = *(const ushort4*)(bG[t] + k0 + 4);
                }
                *(ushort4*)(&Bs[buf][brow[t] * LDA + bcol[t]])     = bv;
                *(ushort4*)(&Bs[buf][brow[t] * LDA + bcol[t] + 4]) = bv2;
            }
        }
    };

    auto compute = [&](int buf) {
        FragBF af[4];
        const int g8 = hi * 8;
#pragma unroll
        for (int i = 0; i < 4; i++) {
            const unsigned short* p = &As[buf][(wm * 64 + i * 16 + r) * LDA];
            af[i].h[0] = *(const v8us*)(p + g8);
            af[i].h[1] = *(const v8us*)(p + 16 + g8);
        }
        FragBF bfr[4];
        const int g16 = hi * 16;
#pragma unroll
        for (int j = 0; j < 4; j++) {
            const unsigned short* p = &Bs[buf][(wn * 64 + j * 16 + r) * LDA + g16 / 2];
            bfr[j].h[0] = *(const v8us*)(p);
            bfr[j].h[1] = *(const v8us*)(p + 16);
        }
#pragma unroll
        for (int i = 0; i < 4; i++)
#pragma unroll
            for (int j = 0; j < 4; j++)
                acc[i][j] = __builtin_amdgcn_wmma_f32_16x16x32_bf16(
                    false, af[i].v, false, bfr[j].v, (short)0, acc[i][j], false, false);
    };

    // prologue: first tile into buf 0
    load_tiles(0, 0);
    wait_async0();
    __syncthreads();

    // K loop unrolled x2 so buffer index is a compile-time constant
    for (int k0 = 0; k0 < K; k0 += 64) {
        const bool last = (k0 + 64 >= K);
        load_tiles(k0 + 32, 1);
        compute(0);
        wait_async0();
        __syncthreads();
        if (!last) load_tiles(k0 + 64, 0);
        compute(1);
        if (!last) wait_async0();
        __syncthreads();
    }

    // straight-line epilogue (variant chosen at compile time)
#pragma unroll
    for (int i = 0; i < 4; i++) {
        const int gm = bm * GBM + wm * 64 + i * 16 + hi * 8;
#pragma unroll
        for (int j = 0; j < 4; j++) {
            const int gn = bn * GBN + wn * 64 + j * 16 + r;
            if (!FULLN && gn >= N) continue;
            const float bv = HASBIAS ? bias[gn] : 0.0f;
            size_t o = (size_t)gm * N + gn;
#pragma unroll
            for (int v = 0; v < 8; v++, o += (size_t)N) {
                float val = acc[i][j][v] + bv;
                if (GELU)   val = 0.5f * val * (1.0f + erff(val * 0.70710678f));
                if (HASRES) val += resid[o];
                if (OUTFP)  outF[o] = val;
                if (OUTBF)  outB[o] = f2bf(val);
            }
        }
    }
}

// ---------------- flash attention (causal), bf16 WMMA ----------------
// block = (q-block of 64 rows, head, batch); 128 thr = 4 waves, 16 q-rows/wave.
// Softmax denominator via ones-matrix WMMA accumulator (o[4]); row max via permlane16.
__global__ __launch_bounds__(128) __attribute__((amdgpu_waves_per_eu(2)))
void attn_flash_wmma(const unsigned short* __restrict__ qkv,
                     unsigned short* __restrict__ obuf)
{
    const int qb = blockIdx.x, h = blockIdx.y, b = blockIdx.z;
    const int tid = threadIdx.x, lane = tid & 31, wave = tid >> 5;
    const int r = lane & 15, hi = lane >> 4;
    const int qrow0 = qb * 64 + wave * 16;

    __shared__ __align__(16) unsigned short Ks[32 * 72];      // [j][d], stride 72
    __shared__ __align__(16) unsigned short Vt[64 * 40];      // [d][j] transposed
    __shared__ __align__(16) unsigned short Ps[4 * 16 * 40];  // per-wave P tile [q][j]

    // Q fragments (A layout), loaded once from global
    FragBF qf[2];
    {
        const int g8 = hi * 8;
        size_t base = ((size_t)(b * TT + qrow0 + r)) * D3 + h * 64;
#pragma unroll
        for (int c = 0; c < 2; c++) {
            qf[c].h[0] = *(const v8us*)(qkv + base + c * 32 + g8);
            qf[c].h[1] = *(const v8us*)(qkv + base + c * 32 + 16 + g8);
        }
    }

    // constant all-ones B fragment (bf16 1.0): row-sum via WMMA
    FragBF onesf;
#pragma unroll
    for (int z = 0; z < 8; z++) { onesf.h[0][z] = 0x3F80; onesf.h[1][z] = 0x3F80; }

    // per-thread K chunk mapping (async): 256 16B chunks, 2 per thread
    int kj[2], kd8[2]; unsigned kL[2];
#pragma unroll
    for (int t = 0; t < 2; t++) {
        int c = tid + t * 128;
        kj[t] = c >> 3; kd8[t] = (c & 7) * 8;
        kL[t] = lds_addr(&Ks[kj[t] * 72 + kd8[t]]);
    }

    float mrow[8];
    v8f o[5] = {};                               // o[0..3] = O tiles, o[4] = row-sum
#pragma unroll
    for (int v = 0; v < 8; v++) mrow[v] = -1e30f;

    const int nkb = qb * 2 + 2;                  // causal: keys up to end of q-block
    for (int kb = 0; kb < nkb; kb++) {
        // K tile via async global->LDS
#pragma unroll
        for (int t = 0; t < 2; t++) {
            const unsigned short* g = qkv +
                ((size_t)(b * TT + kb * 32 + kj[t])) * D3 + DD + h * 64 + kd8[t];
            async_ld16(kL[t], g);
        }
        // V tile (transposed store path)
#pragma unroll
        for (int t = 0; t < 4; t++) {
            int c = tid + t * 128;
            int j = c >> 4;
            int d4 = (c & 15) * 4;
            size_t rowb = ((size_t)(b * TT + kb * 32 + j)) * D3 + h * 64;
            ushort4 vv = *(const ushort4*)(qkv + rowb + 2 * DD + d4);
            Vt[(d4 + 0) * 40 + j] = vv.x;
            Vt[(d4 + 1) * 40 + j] = vv.y;
            Vt[(d4 + 2) * 40 + j] = vv.z;
            Vt[(d4 + 3) * 40 + j] = vv.w;
        }
        wait_async0();
        __syncthreads();

        // S = Q K^T : preload all K fragments, then 4 WMMAs
        FragBF kf[2][2];
        {
            const int g16 = hi * 16;
#pragma unroll
            for (int jt = 0; jt < 2; jt++) {
                const int n = jt * 16 + r;
#pragma unroll
                for (int c = 0; c < 2; c++) {
                    const unsigned short* p = &Ks[n * 72 + c * 32 + g16 / 2];
                    kf[jt][c].h[0] = *(const v8us*)(p);
                    kf[jt][c].h[1] = *(const v8us*)(p + 16);
                }
            }
        }
        v8f s[2];
#pragma unroll
        for (int jt = 0; jt < 2; jt++) {
            v8f z = {};
            z = __builtin_amdgcn_wmma_f32_16x16x32_bf16(false, qf[0].v, false, kf[jt][0].v, (short)0, z, false, false);
            z = __builtin_amdgcn_wmma_f32_16x16x32_bf16(false, qf[1].v, false, kf[jt][1].v, (short)0, z, false, false);
            s[jt] = z;
        }

        // online softmax update (rows live across 16-lane halves in C layout)
        const float scale = 0.125f;              // 1/sqrt(64)
        const int col0 = kb * 32 + r;
#pragma unroll
        for (int v = 0; v < 8; v++) {
            const int q = qrow0 + v + 8 * hi;
            float s0 = s[0][v] * scale;
            float s1 = s[1][v] * scale;
            if (col0 > q)      s0 = -1e30f;
            if (col0 + 16 > q) s1 = -1e30f;
            float mc = rowmax16(fmaxf(s0, s1));
            float mn = fmaxf(mrow[v], mc);
            float cf = __expf(mrow[v] - mn);
            mrow[v] = mn;
            float p0 = __expf(s0 - mn);
            float p1 = __expf(s1 - mn);
#pragma unroll
            for (int dt = 0; dt < 5; dt++) o[dt][v] = o[dt][v] * cf;
            int prow = (wave * 16 + v + 8 * hi) * 40;
            Ps[prow + r]      = f2bf(p0);
            Ps[prow + 16 + r] = f2bf(p1);
        }

        // P fragment (A layout) from this wave's private LDS region
        FragBF pf;
        {
            const int g8 = hi * 8;
            const unsigned short* p = &Ps[(wave * 16 + r) * 40];
            pf.h[0] = *(const v8us*)(p + g8);
            pf.h[1] = *(const v8us*)(p + 16 + g8);
        }
        // preload V fragments, then O += P*V (4 tiles) and rowsum += P*1 (1 tile)
        FragBF vf[4];
        {
            const int g16 = hi * 16;
#pragma unroll
            for (int dt = 0; dt < 4; dt++) {
                const unsigned short* p = &Vt[(dt * 16 + r) * 40 + g16 / 2];
                vf[dt].h[0] = *(const v8us*)(p);
                vf[dt].h[1] = *(const v8us*)(p + 16);
            }
        }
#pragma unroll
        for (int dt = 0; dt < 4; dt++)
            o[dt] = __builtin_amdgcn_wmma_f32_16x16x32_bf16(false, pf.v, false, vf[dt].v, (short)0, o[dt], false, false);
        o[4] = __builtin_amdgcn_wmma_f32_16x16x32_bf16(false, pf.v, false, onesf.v, (short)0, o[4], false, false);
        __syncthreads();
    }

    // normalize by row-sum (replicated across all lanes by the ones-WMMA) and store bf16
#pragma unroll
    for (int dt = 0; dt < 4; dt++)
#pragma unroll
        for (int v = 0; v < 8; v++) {
            const int q = qrow0 + v + 8 * hi;
            float val = o[dt][v] / o[4][v];
            obuf[((size_t)(b * TT + q)) * DD + h * 64 + dt * 16 + r] = f2bf(val);
        }
}

// ---------------- host orchestration ----------------
extern "C" void kernel_launch(void* const* d_in, const int* in_sizes, int n_in,
                              void* d_out, int out_size, void* d_ws, size_t ws_size,
                              hipStream_t stream) {
    const int*   idx    = (const int*)  d_in[0];
    const float* tok    = (const float*)d_in[1];
    const float* pos    = (const float*)d_in[2];
    const float* ln1_w  = (const float*)d_in[3];
    const float* ln1_b  = (const float*)d_in[4];
    const float* qkv_w  = (const float*)d_in[5];
    const float* qkv_b  = (const float*)d_in[6];
    const float* out_w  = (const float*)d_in[7];
    const float* out_b  = (const float*)d_in[8];
    const float* ln2_w  = (const float*)d_in[9];
    const float* ln2_b  = (const float*)d_in[10];
    const float* fc1_w  = (const float*)d_in[11];
    const float* fc1_b  = (const float*)d_in[12];
    const float* fc2_w  = (const float*)d_in[13];
    const float* fc2_b  = (const float*)d_in[14];
    const float* lnf_w  = (const float*)d_in[15];
    const float* lnf_b  = (const float*)d_in[16];
    float* logits = (float*)d_out;

    // workspace carve-up (~196 MB)
    char* base = (char*)d_ws;
    size_t off = 0;
    auto carve = [&](size_t bytes) -> char* {
        char* p = base + off;
        off = (off + bytes + 255) & ~(size_t)255;
        return p;
    };
    float*          x     = (float*)         carve((size_t)BT * DD * 4);
    unsigned short* hbf   = (unsigned short*)carve((size_t)BT * DD * 2);
    unsigned short* qkvbf = (unsigned short*)carve((size_t)BT * D3 * 2);
    unsigned short* obf   = (unsigned short*)carve((size_t)BT * DD * 2);
    unsigned short* ubf   = (unsigned short*)carve((size_t)BT * FF * 2);
    unsigned short* wqkv  = (unsigned short*)carve((size_t)LL * D3 * DD * 2);
    unsigned short* wout  = (unsigned short*)carve((size_t)LL * DD * DD * 2);
    unsigned short* wfc1  = (unsigned short*)carve((size_t)LL * FF * DD * 2);
    unsigned short* wfc2  = (unsigned short*)carve((size_t)LL * DD * FF * 2);
    unsigned short* tokbf = (unsigned short*)carve((size_t)VV * DD * 2);

    // weight conversion f32 -> bf16 (every call; no cached state allowed)
    cvt_f32_bf16<<<4096, 256, 0, stream>>>(qkv_w, wqkv, (long)LL * D3 * DD);
    cvt_f32_bf16<<<4096, 256, 0, stream>>>(out_w, wout, (long)LL * DD * DD);
    cvt_f32_bf16<<<4096, 256, 0, stream>>>(fc1_w, wfc1, (long)LL * FF * DD);
    cvt_f32_bf16<<<4096, 256, 0, stream>>>(fc2_w, wfc2, (long)LL * DD * FF);
    cvt_f32_bf16<<<4096, 256, 0, stream>>>(tok,   tokbf, (long)VV * DD);

    // x = tok_emb[idx] + pos_emb
    embed_kernel<<<BT, 256, 0, stream>>>(idx, tok, pos, x);

    const dim3 gQKV(D3 / GBN + (D3 % GBN ? 1 : 0), BT / GBM);   // 9,16 (exact)
    const dim3 gD  (DD / GBN + (DD % GBN ? 1 : 0), BT / GBM);   // 3,16 (exact)
    const dim3 gF  (FF / GBN + (FF % GBN ? 1 : 0), BT / GBM);   // 12,16 (exact)
    const dim3 gAttn(TT / 64, HH, BB);

    for (int l = 0; l < LL; l++) {
        const unsigned short* wq = wqkv + (size_t)l * D3 * DD;
        const unsigned short* wo = wout + (size_t)l * DD * DD;
        const unsigned short* w1 = wfc1 + (size_t)l * FF * DD;
        const unsigned short* w2 = wfc2 + (size_t)l * DD * FF;

        // h = LN1(x) -> bf16
        ln_bf16_kernel<<<BT, 256, 0, stream>>>(x, ln1_w + (size_t)l * DD, ln1_b + (size_t)l * DD, hbf);
        // qkv = h @ qkv_w.T + qkv_b  (bf16 out)
        gemm_bf16_wmma<true,false,false,false,true,true><<<gQKV, 256, 0, stream>>>(
            hbf, wq, qkv_b + (size_t)l * D3, nullptr, nullptr, qkvbf, BT, D3, DD, 0);
        // o = softmax(QK^T) V  (bf16 out)
        attn_flash_wmma<<<gAttn, 128, 0, stream>>>(qkvbf, obf);
        // x = x + o @ out_w.T + out_b
        gemm_bf16_wmma<true,false,true,true,false,true><<<gD, 256, 0, stream>>>(
            obf, wo, out_b + (size_t)l * DD, x, x, nullptr, BT, DD, DD, 0);
        // h = LN2(x) -> bf16
        ln_bf16_kernel<<<BT, 256, 0, stream>>>(x, ln2_w + (size_t)l * DD, ln2_b + (size_t)l * DD, hbf);
        // u = gelu(h @ fc1_w.T + fc1_b)  (bf16 out)
        gemm_bf16_wmma<true,true,false,false,true,true><<<gF, 256, 0, stream>>>(
            hbf, w1, fc1_b + (size_t)l * FF, nullptr, nullptr, ubf, BT, FF, DD, 0);
        // x = x + u @ fc2_w.T + fc2_b
        gemm_bf16_wmma<true,false,true,true,false,true><<<gD, 256, 0, stream>>>(
            ubf, w2, fc2_b + (size_t)l * DD, x, x, nullptr, BT, DD, FF, 0);
    }

    // final LN + weight-tied head: full blocks async-only, single partial column block separate
    ln_bf16_kernel<<<BT, 256, 0, stream>>>(x, lnf_w, lnf_b, hbf);
    const int headFull = VV / GBN;                               // 196 full N-blocks
    gemm_bf16_wmma<false,false,false,true,false,true><<<dim3(headFull, BT / GBM), 256, 0, stream>>>(
        hbf, tokbf, nullptr, nullptr, logits, nullptr, BT, VV, DD, 0);
    gemm_bf16_wmma<false,false,false,true,false,false><<<dim3(1, BT / GBM), 256, 0, stream>>>(
        hbf, tokbf, nullptr, nullptr, logits, nullptr, BT, VV, DD, headFull);
    (void)in_sizes; (void)n_in; (void)out_size; (void)ws_size;
}
